// RelativeMultiheadAttention_25297357373941
// MI455X (gfx1250) — matrix operand order repaired and checked by reading, compiled, not run on hardware
//
#include <hip/hip_runtime.h>
#include <hip/hip_bf16.h>

// ---------------------------------------------------------------------------
// RelativeMultiheadAttention for MI455X (gfx1250), wave32, bf16 WMMA + f32 acc
// GEMMs: 128x128 macro-tiles, double-buffered LDS staging via Tensor Data
// Mover (tensor_load_to_lds + s_wait_tensorcnt).
// ---------------------------------------------------------------------------
// T=1024, B=4, E=1024, H=16, D=64, L=16 (33 relative positions)

#define T_DIM 1024
#define B_DIM 4
#define E_DIM 1024
#define H_DIM 16
#define D_DIM 64
#define BH    (B_DIM * H_DIM)   // 64
#define ROWS  (T_DIM * B_DIM)   // 4096
#define NREL  33

typedef __attribute__((ext_vector_type(16))) __bf16    v16bf;
typedef __attribute__((ext_vector_type(8)))  float     v8f;
typedef __attribute__((ext_vector_type(4)))  unsigned  v4u;
typedef __attribute__((ext_vector_type(8)))  int       v8i_;
typedef __attribute__((ext_vector_type(4)))  int       v4i_;

// ---------------- scalar bf16 helpers -------------------------------------
__device__ __forceinline__ unsigned short f2bf(float f) {
    union { float f; unsigned u; } v; v.f = f;
    unsigned r = v.u + 0x7FFFu + ((v.u >> 16) & 1u);   // round-to-nearest-even
    return (unsigned short)(r >> 16);
}
__device__ __forceinline__ float bf2f(unsigned short h) {
    union { unsigned u; float f; } v; v.u = ((unsigned)h) << 16;
    return v.f;
}

// ---------------- WMMA wrapper ---------------------------------------------
__device__ __forceinline__ v8f wmma_bf16(v16bf a, v16bf b, v8f c) {
    return __builtin_amdgcn_wmma_f32_16x16x32_bf16(
        /*neg_a=*/false, a, /*neg_b=*/false, b,
        /*c_mod=*/(short)0, c, /*reuse_a=*/false, /*reuse_b=*/false);
}

// A fragment (16x32 bf16, M x K) from row-major [*, lda] bf16 matrix.
// Lane L holds row M = m0 + (L&15); K pair base = k0 + (L>=16 ? 8 : 0);
// VGPR v holds K = kb + (v&3)*2 + (v>=4 ? 16 : 0) .. +1
__device__ __forceinline__ v16bf load_A_frag(const unsigned short* A, int lda,
                                             int m0, int k0, int lane) {
    int M  = m0 + (lane & 15);
    int kb = k0 + ((lane >> 4) << 3);
    const unsigned short* p = A + M * lda;
    union { v16bf v; unsigned u[8]; } r;
#pragma unroll
    for (int v = 0; v < 8; ++v) {
        int K = kb + ((v & 3) << 1) + ((v >> 2) << 4);
        r.u[v] = *(const unsigned*)(p + K);
    }
    return r.v;
}

// B fragment (32x16 bf16, K x N) where memory holds the [N, K] row-major
// matrix (weights stored [out,in], K stored [s,d], V stored transposed [d,t]).
// Lane L holds col N = n0 + (L&15); K = k0 + (L>=16 ? 8 : 0) + (v&3)*2 + (v>=4 ? 16 : 0) .. +1
__device__ __forceinline__ v16bf load_B_fragT(const unsigned short* Bm, int ldb,
                                              int n0, int k0, int lane) {
    int N  = n0 + (lane & 15);
    int kb = k0 + ((lane >> 4) << 3);
    const unsigned short* p = Bm + N * ldb + kb;
    union { v16bf v; unsigned u[8]; } r;
#pragma unroll
    for (int v = 0; v < 8; ++v) r.u[v] = *(const unsigned*)(p + ((v & 3) << 1) + ((v >> 2) << 4));
    return r.v;
}

// ---------------- Tensor Data Mover: 2-D tile -> LDS -----------------------
// Loads a (rows x 32 bf16) tile from a row-major bf16 matrix into LDS,
// packed contiguously as [rows][32] bf16. Descriptor per CDNA5 ISA.
// Must be called from a wave with uniform arguments.
__device__ __forceinline__ void tdm_load_tile32(const unsigned short* gsrc,
                                                int row_pitch_elems,  // bf16/row
                                                unsigned lds_byte_off,
                                                unsigned rows) {
    unsigned long long ga = (unsigned long long)gsrc;
    v4u g0;
    g0[0] = 1u;                                   // count=1, user descriptor
    g0[1] = lds_byte_off;                         // lds_addr (bytes)
    g0[2] = (unsigned)ga;                         // global_addr[31:0]
    g0[3] = (unsigned)((ga >> 32) & 0x01FFFFFFu)  // global_addr[56:32]
            | (2u << 30);                         // type=2 ("image")
    unsigned dim0   = (unsigned)(row_pitch_elems >> 1);  // tensor dim0 (u32 units)
    unsigned dim1   = 1u << 20;                          // tall enough
    unsigned tile0  = 16u;                               // 32 bf16 = 16 u32
    unsigned tile1  = rows;
    unsigned long long s0 = (unsigned long long)(row_pitch_elems >> 1);
    v8i_ g1;
    g1[0] = (int)(2u << 16);                      // data_size = 4 bytes
    g1[1] = (int)((dim0 & 0xFFFFu) << 16);        // dim0[15:0] -> bits[63:48]
    g1[2] = (int)((dim0 >> 16) | ((dim1 & 0xFFFFu) << 16));
    g1[3] = (int)((dim1 >> 16) | (tile0 << 16));  // tile_dim0 -> bits[127:112]
    g1[4] = (int)(tile1 & 0xFFFFu);               // tile_dim1; tile_dim2=0
    g1[5] = (int)(unsigned)(s0 & 0xFFFFFFFFu);    // dim0_stride[31:0]
    g1[6] = (int)(unsigned)((s0 >> 32) & 0xFFFFu);// dim0_stride[47:32]
    g1[7] = 0;                                    // dim1_stride unused (2-D)
    v4i_ gz = {0, 0, 0, 0};
#if defined(__clang_major__) && (__clang_major__ >= 23)
    v8i_ gz8 = {0, 0, 0, 0, 0, 0, 0, 0};
    __builtin_amdgcn_tensor_load_to_lds(g0, g1, gz, gz, gz8, 0);
#else
    __builtin_amdgcn_tensor_load_to_lds(g0, g1, gz, gz, 0);
#endif
}

// ---------------- kernels ---------------------------------------------------

__global__ void cvt_f32_bf16_kernel(const float* __restrict__ in,
                                    unsigned short* __restrict__ out, int n) {
    int i = blockIdx.x * blockDim.x + threadIdx.x;
    if (i < n) out[i] = f2bf(in[i]);
}

// Macro-tile GEMM body: 8 waves, C tile 128(M) x 128(N), K step 32,
// double-buffered TDM staging: tiles for step i+1 stream in behind the
// WMMA burst for step i; s_wait_tensorcnt(2) releases only the older pair
// (TDM completions are in-order per issuing wave).
// tA/tB: [128][32] bf16 LDS buffers (x2).
__device__ __forceinline__ void gemm_128x128_bf16(
        const unsigned short* __restrict__ A, int lda, int m0,
        const unsigned short* __restrict__ W, int ldw, int n0,
        int Kdim,
        unsigned short* tA0, unsigned short* tB0,
        unsigned short* tA1, unsigned short* tB1, v8f acc[8]) {
    int lane = threadIdx.x & 31;
    int mloc = (threadIdx.x >> 5) << 4;       // wave*16
    unsigned short* tA[2] = {tA0, tA1};
    unsigned short* tB[2] = {tB0, tB1};
    unsigned ldsA[2] = {(unsigned)(unsigned long long)tA0,
                        (unsigned)(unsigned long long)tA1};
    unsigned ldsB[2] = {(unsigned)(unsigned long long)tB0,
                        (unsigned)(unsigned long long)tB1};
    const int steps = Kdim / 32;
    if (threadIdx.x < 32) {                    // wave 0 issues all TDM loads
        tdm_load_tile32(A + m0 * lda, lda, ldsA[0], 128u);
        tdm_load_tile32(W + n0 * ldw, ldw, ldsB[0], 128u);
    }
    for (int i = 0; i < steps; ++i) {
        int cur = i & 1, nxt = cur ^ 1;
        if (i + 1 < steps) {
            if (threadIdx.x < 32) {            // prefetch next K slice
                int k0 = (i + 1) * 32;
                tdm_load_tile32(A + m0 * lda + k0, lda, ldsA[nxt], 128u);
                tdm_load_tile32(W + n0 * ldw + k0, ldw, ldsB[nxt], 128u);
            }
            __builtin_amdgcn_s_wait_tensorcnt(2);  // older pair done
        } else {
            __builtin_amdgcn_s_wait_tensorcnt(0);
        }
        __syncthreads();                       // current tiles visible to all
        v16bf a = load_A_frag(tA[cur], 32, mloc, 0, lane);
#pragma unroll
        for (int nt = 0; nt < 8; ++nt) {
            v16bf b = load_B_fragT(tB[cur], 32, nt * 16, 0, lane);
            acc[nt] = wmma_bf16(a, b, acc[nt]);
        }
        __syncthreads();                       // done with buf before reuse
    }
}

// QKV = X[4096,1024] * W[3072,1024]^T + bias ; scatter into attention layouts.
// grid (3072/128, 4096/128), block 256 (8 waves)
__global__ __launch_bounds__(256)
void qkv_wmma_kernel(const unsigned short* __restrict__ X,
                     const unsigned short* __restrict__ W,
                     const float* __restrict__ bias,
                     unsigned short* __restrict__ Qb,   // [bh][t][d] (pre-scaled)
                     unsigned short* __restrict__ Kb,   // [bh][s][d]
                     unsigned short* __restrict__ Vt) { // [bh][d][t]
    __shared__ unsigned short tA0[128 * 32], tB0[128 * 32];
    __shared__ unsigned short tA1[128 * 32], tB1[128 * 32];
    int lane = threadIdx.x & 31;
    int mloc = (threadIdx.x >> 5) << 4;
    int n0 = blockIdx.x * 128;
    int m0 = blockIdx.y * 128;
    v8f acc[8] = {};
    gemm_128x128_bf16(X, E_DIM, m0, W, E_DIM, n0, E_DIM, tA0, tB0, tA1, tB1, acc);
#pragma unroll
    for (int nt = 0; nt < 8; ++nt) {
        int N = n0 + nt * 16 + (lane & 15);
        float bval = bias[N];
        int sec = N >> 10;                     // 0=q 1=k 2=v
        int c = N & 1023;
        int h = c >> 6, d = c & 63;
#pragma unroll
        for (int r = 0; r < 8; ++r) {
            int M = m0 + mloc + r + ((lane >> 4) << 3);
            int t = M >> 2, bI = M & 3;        // row = t*B + b
            int bh = bI * H_DIM + h;
            float val = acc[nt][r] + bval;
            if (sec == 0)
                Qb[(bh * T_DIM + t) * D_DIM + d] = f2bf(val * 0.125f); // D^-0.5
            else if (sec == 1)
                Kb[(bh * T_DIM + t) * D_DIM + d] = f2bf(val);
            else
                Vt[(bh * D_DIM + d) * T_DIM + t] = f2bf(val);
        }
    }
}

// qrel[bh][t][r] = sum_d Qb[bh,t,d] * rel_pos_keys[r, d]   (d < 64 of E=1024)
__global__ void qrel_kernel(const unsigned short* __restrict__ Qb,
                            const float* __restrict__ relk,
                            float* __restrict__ qrel) {
    int idx = blockIdx.x * blockDim.x + threadIdx.x;
    if (idx >= BH * T_DIM * NREL) return;
    int r  = idx % NREL;
    int t  = (idx / NREL) % T_DIM;
    int bh = idx / (NREL * T_DIM);
    const unsigned short* q = Qb + (bh * T_DIM + t) * D_DIM;
    const float* rk = relk + r * E_DIM;
    float s = 0.f;
#pragma unroll 8
    for (int d = 0; d < D_DIM; ++d) s += bf2f(q[d]) * rk[d];
    qrel[idx] = s;
}

// Attention: one workgroup (4 waves) per (bh, 16-row q tile).
// Scores (16x1024 f32 = 64KB) live in LDS; probs packed bf16 in place.
// grid (T/16, BH), block 128
__global__ __launch_bounds__(128)
void attn_wmma_kernel(const unsigned short* __restrict__ Qb,
                      const unsigned short* __restrict__ Kb,
                      const unsigned short* __restrict__ Vt,
                      const float* __restrict__ qrel,
                      unsigned short* __restrict__ Obf) { // [t*B+b][e]
    __shared__ float sc[16 * T_DIM];           // 64 KB (of 320 KB WGP pool)
    int bh   = blockIdx.y;
    int q0   = blockIdx.x * 16;
    int wave = threadIdx.x >> 5;
    int lane = threadIdx.x & 31;

    const unsigned short* Qh = Qb + bh * T_DIM * D_DIM;
    const unsigned short* Kh = Kb + bh * T_DIM * D_DIM;
    const unsigned short* Vh = Vt + bh * D_DIM * T_DIM;

    // ---- phase A: scores = (Q K^T) tile + qrel --------------------------
    for (int st = wave; st < T_DIM / 16; st += 4) {
        v8f c = {};
#pragma unroll
        for (int k0 = 0; k0 < D_DIM; k0 += 32) {
            v16bf a = load_A_frag (Qh, D_DIM, q0,      k0, lane);
            v16bf b = load_B_fragT(Kh, D_DIM, st * 16, k0, lane);
            c = wmma_bf16(a, b, c);
        }
        int s = st * 16 + (lane & 15);
#pragma unroll
        for (int r = 0; r < 8; ++r) {
            int m = r + ((lane >> 4) << 3);
            int t = q0 + m;
            int dd = s - t;
            dd = dd < -16 ? -16 : (dd > 16 ? 16 : dd);
            sc[m * T_DIM + s] = c[r] + qrel[(bh * T_DIM + t) * NREL + dd + 16];
        }
    }
    __syncthreads();

    // ---- phase B: row softmax, pack probs as bf16 pairs in place --------
    if (threadIdx.x < 16) {
        float* row = sc + threadIdx.x * T_DIM;
        float mx = -3.4e38f;
        for (int s = 0; s < T_DIM; ++s) mx = fmaxf(mx, row[s]);
        float sum = 0.f;
        for (int s = 0; s < T_DIM; ++s) { float e = __expf(row[s] - mx); row[s] = e; sum += e; }
        float inv = 1.0f / sum;
        unsigned* rowu = (unsigned*)row;       // pack into first half of own row
        for (int s = 0; s < T_DIM; s += 2) {
            unsigned short p0 = f2bf(row[s]     * inv);
            unsigned short p1 = f2bf(row[s + 1] * inv);
            rowu[s >> 1] = (unsigned)p0 | ((unsigned)p1 << 16);
        }
    }
    __syncthreads();

    // ---- phase C: O = P V ; wave w owns d-tile [w*16, w*16+16) ----------
    {
        int nt = wave;
        v8f o = {};
        for (int k0 = 0; k0 < T_DIM; k0 += 32) {
            // A fragment from packed bf16 probs in LDS
            int M  = lane & 15;
            int kb = k0 + ((lane >> 4) << 3);
            const unsigned* rowu = (const unsigned*)(sc + M * T_DIM);
            union { v16bf v; unsigned u[8]; } a;
#pragma unroll
            for (int v = 0; v < 8; ++v) {
                int K = kb + ((v & 3) << 1) + ((v >> 2) << 4);
                a.u[v] = rowu[K >> 1];
            }
            v16bf b = load_B_fragT(Vh, T_DIM, nt * 16, k0, lane);
            o = wmma_bf16(a.v, b, o);
        }
        int n = lane & 15;
        int b_ = bh >> 4, h = bh & 15;
#pragma unroll
        for (int r = 0; r < 8; ++r) {
            int m = r + ((lane >> 4) << 3);
            int t = q0 + m;
            int e = h * D_DIM + nt * 16 + n;
            Obf[(t * B_DIM + b_) * E_DIM + e] = f2bf(o[r]);
        }
    }
}

// out[row, n] = sum_e Obf[row,e] * out_w[n,e] + out_b[n]  (f32 output)
// grid (1024/128, 4096/128), block 256
__global__ __launch_bounds__(256)
void outproj_wmma_kernel(const unsigned short* __restrict__ A,
                         const unsigned short* __restrict__ W,
                         const float* __restrict__ bias,
                         float* __restrict__ C) {
    __shared__ unsigned short tA0[128 * 32], tB0[128 * 32];
    __shared__ unsigned short tA1[128 * 32], tB1[128 * 32];
    int lane = threadIdx.x & 31;
    int mloc = (threadIdx.x >> 5) << 4;
    int n0 = blockIdx.x * 128;
    int m0 = blockIdx.y * 128;
    v8f acc[8] = {};
    gemm_128x128_bf16(A, E_DIM, m0, W, E_DIM, n0, E_DIM, tA0, tB0, tA1, tB1, acc);
#pragma unroll
    for (int nt = 0; nt < 8; ++nt) {
        int N = n0 + nt * 16 + (lane & 15);
        float bv = bias[N];
#pragma unroll
        for (int r = 0; r < 8; ++r) {
            int M = m0 + mloc + r + ((lane >> 4) << 3);
            C[M * E_DIM + N] = acc[nt][r] + bv;
        }
    }
}

// ---------------------------------------------------------------------------
extern "C" void kernel_launch(void* const* d_in, const int* in_sizes, int n_in,
                              void* d_out, int out_size, void* d_ws, size_t ws_size,
                              hipStream_t stream) {
    const float* query   = (const float*)d_in[0];   // [T,B,E]
    const float* ipw     = (const float*)d_in[1];   // [3E,E]
    const float* ipb     = (const float*)d_in[2];   // [3E]
    const float* relk    = (const float*)d_in[3];   // [33,E]
    const float* outw    = (const float*)d_in[4];   // [E,E]
    const float* outb    = (const float*)d_in[5];   // [E]
    float* out = (float*)d_out;                     // [T,B,E]

    // workspace layout (bytes)
    char* ws = (char*)d_ws;
    size_t off = 0;
    auto alloc = [&](size_t bytes) { char* p = ws + off; off += (bytes + 255) & ~size_t(255); return p; };
    unsigned short* Xbf  = (unsigned short*)alloc((size_t)ROWS * E_DIM * 2);      // 8 MB
    unsigned short* Wbf  = (unsigned short*)alloc((size_t)3 * E_DIM * E_DIM * 2); // 6 MB
    unsigned short* OWbf = (unsigned short*)alloc((size_t)E_DIM * E_DIM * 2);     // 2 MB
    unsigned short* Qbf  = (unsigned short*)alloc((size_t)BH * T_DIM * D_DIM * 2);
    unsigned short* Kbf  = (unsigned short*)alloc((size_t)BH * T_DIM * D_DIM * 2);
    unsigned short* Vtb  = (unsigned short*)alloc((size_t)BH * D_DIM * T_DIM * 2);
    float*          qrel = (float*)alloc((size_t)BH * T_DIM * NREL * 4);          // 8.6 MB
    unsigned short* Obf  = (unsigned short*)alloc((size_t)ROWS * E_DIM * 2);      // 8 MB
    (void)ws_size; (void)in_sizes; (void)n_in; (void)out_size;

    // 1) f32 -> bf16 conversions
    {
        int n = ROWS * E_DIM;
        cvt_f32_bf16_kernel<<<(n + 255) / 256, 256, 0, stream>>>(query, Xbf, n);
        n = 3 * E_DIM * E_DIM;
        cvt_f32_bf16_kernel<<<(n + 255) / 256, 256, 0, stream>>>(ipw, Wbf, n);
        n = E_DIM * E_DIM;
        cvt_f32_bf16_kernel<<<(n + 255) / 256, 256, 0, stream>>>(outw, OWbf, n);
    }
    // 2) fused QKV projection (WMMA, double-buffered TDM 128x128 tiles)
    qkv_wmma_kernel<<<dim3(3 * E_DIM / 128, ROWS / 128), 256, 0, stream>>>(
        Xbf, Wbf, ipb, Qbf, Kbf, Vtb);
    // 3) relative-position dot products (33 distinct offsets)
    {
        int n = BH * T_DIM * NREL;
        qrel_kernel<<<(n + 255) / 256, 256, 0, stream>>>(Qbf, relk, qrel);
    }
    // 4) attention (scores + rel bias + softmax + PV), WMMA
    attn_wmma_kernel<<<dim3(T_DIM / 16, BH), 128, 0, stream>>>(
        Qbf, Kbf, Vtb, qrel, Obf);
    // 5) output projection (WMMA, double-buffered TDM) + bias -> f32 out
    outproj_wmma_kernel<<<dim3(E_DIM / 128, ROWS / 128), 256, 0, stream>>>(
        Obf, OWbf, outb, out);
}
